// SlotAttentionNoGRU_16286515986740
// MI455X (gfx1250) — matrix operand order repaired and checked by reading, not compiled, not run on hardware
//
#include <hip/hip_runtime.h>
#include <hip/hip_bf16.h>
#include <math.h>

// ---------------------------------------------------------------------------
// Slot-attention tracker for MI455X (gfx1250, wave32, WMMA).
// Heavy GEMMs (65536x256x256) run on v_wmma_f32_16x16x32_f16 (f16 in, f32 acc).
// Tiny K=16 slot math (GRU/MLP/LN/match) runs in f32 VALU single-workgroup
// kernels. All reductions deterministic (no float atomics).
// ---------------------------------------------------------------------------

typedef __attribute__((ext_vector_type(16))) _Float16 v16h;
typedef __attribute__((ext_vector_type(8)))  _Float16 v8h;
typedef __attribute__((ext_vector_type(8)))  float    v8f;

#define NDET   65536
#define HD     256
#define KSLOT  16
#define SPLITK 64          // split-K groups for the updates GEMM
#define NSPART 4096        // NDET/16 = per-wave slot-sum partials

// ---- WMMA fragment helpers (CDNA5 16-bit A/B layouts, ISA 7.12.2) ----------
// A (16xK row-major, per lane: row = lane%16, K-chunks depend on lane>>4):
//   lanes 0-15: halfs k0+{0..7} and k0+{16..23}; lanes 16-31: +8.
__device__ __forceinline__ v16h load_a_frag(const _Float16* __restrict__ row,
                                            int k0, int hi) {
  v8h lo = *reinterpret_cast<const v8h*>(row + k0 + hi * 8);
  v8h hh = *reinterpret_cast<const v8h*>(row + k0 + hi * 8 + 16);
  return __builtin_shufflevector(lo, hh, 0,1,2,3,4,5,6,7,8,9,10,11,12,13,14,15);
}
// B (Kx16): stored transposed Bt[col][k]; lane col = lane%16,
//   lanes 0-15: halfs k0+{0..7} and k0+{16..23}; lanes 16-31: +8 (same as A).
__device__ __forceinline__ v16h load_b_frag(const _Float16* __restrict__ colrow,
                                            int k0, int hi) {
  v8h lo = *reinterpret_cast<const v8h*>(colrow + k0 + hi * 8);
  v8h hh = *reinterpret_cast<const v8h*>(colrow + k0 + hi * 8 + 16);
  return __builtin_shufflevector(lo, hh, 0,1,2,3,4,5,6,7,8,9,10,11,12,13,14,15);
}
__device__ __forceinline__ v8f wmma_f16f32(v16h a, v16h b, v8f c) {
  return __builtin_amdgcn_wmma_f32_16x16x32_f16(false, a, false, b,
                                                (short)0, c, false, false);
}

// ---------------------------------------------------------------------------
// f32 -> f16 conversion of activations
__global__ __launch_bounds__(256) void cvt_f32_f16(const float* __restrict__ x,
                                                   _Float16* __restrict__ y,
                                                   size_t n) {
  size_t i = (size_t)blockIdx.x * 256 + threadIdx.x;
  if (i < n) y[i] = (_Float16)x[i];
}

// Convert+transpose the 4 big weight matrices (det_W1, det_W2, Wk, Wv) to
// f16 Bt layout: out[w][n*256 + k] = W[k*256 + n].
__global__ __launch_bounds__(256) void prep_weights(
    const float* __restrict__ W1, const float* __restrict__ W2,
    const float* __restrict__ Wk, const float* __restrict__ Wv,
    _Float16* __restrict__ out) {
  int idx = blockIdx.x * 256 + threadIdx.x;   // 4*65536 total
  int w = idx >> 16, e = idx & 65535;
  int n = e >> 8, k = e & 255;
  const float* src = (w == 0) ? W1 : (w == 1) ? W2 : (w == 2) ? Wk : Wv;
  out[idx] = (_Float16)src[k * 256 + n];
}

// ---------------------------------------------------------------------------
// Main WMMA GEMM: C[M x 256] = act(A[M x 256] @ B[256 x 256] + bias)
// A: f16 row-major. Bt: f16 transposed [n][k]. 8 waves/block, wave = 16x64 tile.
// OUTMODE: 0 = f32 row-major, 1 = f16 row-major, 2 = f16 transposed (ldT cols).
template <int OUTMODE, bool RELU, bool BIAS>
__global__ __launch_bounds__(256) void gemm256(
    const _Float16* __restrict__ A, const _Float16* __restrict__ Bt,
    const float* __restrict__ bias, void* __restrict__ Cout, int ldT) {
  const int lane = threadIdx.x & 31;
  const int wave = threadIdx.x >> 5;
  const int hi   = lane >> 4;
  const int mrow = blockIdx.x * 64 + (wave & 3) * 16;
  const int colb = blockIdx.y * 128 + (wave >> 2) * 64;

  v8f acc[4];
#pragma unroll
  for (int t = 0; t < 4; ++t)
#pragma unroll
    for (int r = 0; r < 8; ++r) acc[t][r] = 0.0f;

  const _Float16* arow = A + (size_t)(mrow + (lane & 15)) * HD;
#pragma unroll
  for (int k0 = 0; k0 < HD; k0 += 32) {
    v16h a = load_a_frag(arow, k0, hi);
#pragma unroll
    for (int t = 0; t < 4; ++t) {
      const _Float16* brow = Bt + (size_t)(colb + t * 16 + (lane & 15)) * HD;
      v16h b = load_b_frag(brow, k0, hi);
      acc[t] = wmma_f16f32(a, b, acc[t]);
    }
  }

  const int r0 = mrow + hi * 8;
#pragma unroll
  for (int t = 0; t < 4; ++t) {
    const int col = colb + t * 16 + (lane & 15);
    const float bv = BIAS ? bias[col] : 0.0f;
    if (OUTMODE == 0) {
      float* C = (float*)Cout;
#pragma unroll
      for (int r = 0; r < 8; ++r) {
        float v = acc[t][r] + bv;
        if (RELU) v = fmaxf(v, 0.0f);
        C[(size_t)(r0 + r) * HD + col] = v;
      }
    } else if (OUTMODE == 1) {
      _Float16* C = (_Float16*)Cout;
#pragma unroll
      for (int r = 0; r < 8; ++r) {
        float v = acc[t][r] + bv;
        if (RELU) v = fmaxf(v, 0.0f);
        C[(size_t)(r0 + r) * HD + col] = (_Float16)v;
      }
    } else {  // transposed f16: C[col][r0..r0+7], one 16B store
      v8h hv;
#pragma unroll
      for (int r = 0; r < 8; ++r) {
        float v = acc[t][r] + bv;
        if (RELU) v = fmaxf(v, 0.0f);
        hv[r] = (_Float16)v;
      }
      *reinterpret_cast<v8h*>((_Float16*)Cout + (size_t)col * ldT + r0) = hv;
    }
  }
}

// ---------------------------------------------------------------------------
// Row layernorm: y_f16[row] = LN(x_f32[row]) * g + b.  One wave per row.
__global__ __launch_bounds__(256) void layernorm_rows(
    const float* __restrict__ x, _Float16* __restrict__ y,
    const float* __restrict__ g, const float* __restrict__ b) {
  const int lane = threadIdx.x & 31;
  const int row  = blockIdx.x * 8 + (threadIdx.x >> 5);
  const float* xr = x + (size_t)row * HD;
  float v[8], s = 0.0f, sq = 0.0f;
#pragma unroll
  for (int j = 0; j < 8; ++j) {
    v[j] = xr[lane * 8 + j];
    s += v[j];
    sq += v[j] * v[j];
  }
#pragma unroll
  for (int m = 16; m >= 1; m >>= 1) {
    s  += __shfl_xor(s, m, 32);
    sq += __shfl_xor(sq, m, 32);
  }
  const float mu = s * (1.0f / HD);
  const float rs = rsqrtf(sq * (1.0f / HD) - mu * mu + 1e-5f);
#pragma unroll
  for (int j = 0; j < 8; ++j) {
    int c = lane * 8 + j;
    y[(size_t)row * HD + c] = (_Float16)((v[j] - mu) * rs * g[c] + b[c]);
  }
}

// ---------------------------------------------------------------------------
// slots = mu + exp(log_sigma) * noise   (16x256)
__global__ __launch_bounds__(256) void slots_init(
    float* __restrict__ slots, const float* __restrict__ mu,
    const float* __restrict__ lsig, const float* __restrict__ noise) {
  int t = threadIdx.x;
  for (int i = 0; i < KSLOT; ++i) {
    int e = i * HD + t;
    slots[e] = mu[t] + __expf(lsig[t]) * noise[e];
  }
}

// q_f16 = (LN(slots; ln_s) @ Wq) * H^-0.5   (16x256, tiny, single block)
__global__ __launch_bounds__(256) void slot_q_kernel(
    const float* __restrict__ slots, const float* __restrict__ g,
    const float* __restrict__ b, const float* __restrict__ Wq,
    _Float16* __restrict__ q16, float scale) {
  __shared__ float sb[KSLOT][HD];
  __shared__ float st[KSLOT][2];
  const int t = threadIdx.x;
  for (int k = 0; k < KSLOT; ++k) sb[k][t] = slots[k * HD + t];
  __syncthreads();
  if (t < KSLOT) {
    float mu = 0.0f;
    for (int j = 0; j < HD; ++j) mu += sb[t][j];
    mu *= (1.0f / HD);
    float var = 0.0f;
    for (int j = 0; j < HD; ++j) { float d = sb[t][j] - mu; var += d * d; }
    st[t][0] = mu;
    st[t][1] = rsqrtf(var * (1.0f / HD) + 1e-5f);
  }
  __syncthreads();
  for (int k = 0; k < KSLOT; ++k)
    sb[k][t] = (sb[k][t] - st[k][0]) * st[k][1] * g[t] + b[t];
  __syncthreads();
  for (int k = 0; k < KSLOT; ++k) {
    float a = 0.0f;
    for (int j = 0; j < HD; ++j) a += sb[k][j] * Wq[j * HD + t];
    q16[k * HD + t] = (_Float16)(a * scale);
  }
}

// ---------------------------------------------------------------------------
// logits[k,n] = q @ k^T (WMMA), softmax over k in-registers, write p_f16[k][n]
// and deterministic per-wave slot-sum partials spart[waveId][k].
__global__ __launch_bounds__(256) void logits_softmax(
    const _Float16* __restrict__ q, const _Float16* __restrict__ kmat,
    _Float16* __restrict__ p, float* __restrict__ spart) {
  const int lane = threadIdx.x & 31;
  const int wave = threadIdx.x >> 5;
  const int hi   = lane >> 4;
  const int nb   = blockIdx.x * 128 + wave * 16;

  const _Float16* arow = q + (size_t)(lane & 15) * HD;
  const _Float16* brow = kmat + (size_t)(nb + (lane & 15)) * HD;
  v8f acc;
#pragma unroll
  for (int r = 0; r < 8; ++r) acc[r] = 0.0f;
#pragma unroll
  for (int k0 = 0; k0 < HD; k0 += 32) {
    v16h a = load_a_frag(arow, k0, hi);
    v16h b = load_b_frag(brow, k0, hi);
    acc = wmma_f16f32(a, b, acc);
  }

  // softmax over the 16 slots of column n: my 8 rows + partner lane's 8 rows
  float mx = acc[0];
#pragma unroll
  for (int r = 1; r < 8; ++r) mx = fmaxf(mx, acc[r]);
  mx = fmaxf(mx, __shfl_xor(mx, 16, 32));
  float ev[8], s = 0.0f;
#pragma unroll
  for (int r = 0; r < 8; ++r) { ev[r] = __expf(acc[r] - mx); s += ev[r]; }
  s += __shfl_xor(s, 16, 32);
  const float inv = 1.0f / s;

  const int n = nb + (lane & 15);
  float pv[8];
#pragma unroll
  for (int r = 0; r < 8; ++r) {
    pv[r] = ev[r] * inv;
    p[(size_t)(hi * 8 + r) * NDET + n] = (_Float16)pv[r];
  }
  // per-row sums across the 16 columns of this wave (xor stays within half)
#pragma unroll
  for (int m = 1; m < 16; m <<= 1)
#pragma unroll
    for (int r = 0; r < 8; ++r) pv[r] += __shfl_xor(pv[r], m, 32);
  if ((lane & 15) == 0) {
    const int wid = blockIdx.x * 8 + wave;
#pragma unroll
    for (int r = 0; r < 8; ++r) spart[wid * KSLOT + hi * 8 + r] = pv[r];
  }
}

// ---------------------------------------------------------------------------
// updates_raw partials: updpart[s][16][256] = p[16 x Nchunk] @ v[Nchunk x 256]
// split-K over N; LDS cross-wave reduce inside each block.
__global__ __launch_bounds__(256) void updates_gemm(
    const _Float16* __restrict__ p, const _Float16* __restrict__ vt,
    float* __restrict__ updpart) {
  __shared__ float red[8][32][8];
  const int lane = threadIdx.x & 31;
  const int wave = threadIdx.x >> 5;
  const int hi   = lane >> 4;
  const int hb   = blockIdx.x * 16;                       // h column tile
  const int kw   = blockIdx.y * (NDET / SPLITK) + wave * 128;

  const _Float16* arow = p + (size_t)(lane & 15) * NDET;  // slot row
  const _Float16* brow = vt + (size_t)(hb + (lane & 15)) * NDET;
  v8f acc;
#pragma unroll
  for (int r = 0; r < 8; ++r) acc[r] = 0.0f;
#pragma unroll
  for (int s = 0; s < 4; ++s) {
    const int k0 = kw + s * 32;
    v16h a = load_a_frag(arow, k0, hi);
    v16h b = load_b_frag(brow, k0, hi);
    acc = wmma_f16f32(a, b, acc);
  }
#pragma unroll
  for (int r = 0; r < 8; ++r) red[wave][lane][r] = acc[r];
  __syncthreads();
  if (wave == 0) {
    const int col = hb + (lane & 15);
#pragma unroll
    for (int r = 0; r < 8; ++r) {
      float t = 0.0f;
#pragma unroll
      for (int w = 0; w < 8; ++w) t += red[w][lane][r];
      updpart[((size_t)blockIdx.y * KSLOT + hi * 8 + r) * HD + col] = t;
    }
  }
}

// ---------------------------------------------------------------------------
// Per-iteration slot update: reduce partials, normalize, GRU cell, LN + MLP.
__global__ __launch_bounds__(256) void slot_update(
    float* __restrict__ slots, const float* __restrict__ updpart,
    const float* __restrict__ spart,
    const float* __restrict__ W_ih, const float* __restrict__ W_hh,
    const float* __restrict__ b_ih, const float* __restrict__ b_hh,
    const float* __restrict__ ln_g, const float* __restrict__ ln_b,
    const float* __restrict__ mW1, const float* __restrict__ mb1,
    const float* __restrict__ mW2, const float* __restrict__ mb2) {
  __shared__ float bufA[KSLOT][HD];   // updates, later h_ln
  __shared__ float bufB[KSLOT][HD];   // old slots, later mlp hidden
  __shared__ float snew[KSLOT][HD];
  __shared__ float ssum[KSLOT];
  __shared__ float st[KSLOT][2];
  const int t = threadIdx.x;

  if (t < KSLOT) {
    float s = 0.0f;
    for (int w = 0; w < NSPART; ++w) s += spart[w * KSLOT + t];
    ssum[t] = s;
  }
  for (int k = 0; k < KSLOT; ++k) {
    const int e = k * HD + t;
    bufB[k][t] = slots[e];
    float u = 0.0f;
    for (int s = 0; s < SPLITK; ++s) u += updpart[(size_t)s * KSLOT * HD + e];
    bufA[k][t] = u;
  }
  __syncthreads();
  for (int k = 0; k < KSLOT; ++k) bufA[k][t] = bufA[k][t] / (ssum[k] + 1e-8f);
  __syncthreads();

  // GRUCell(updates=bufA, hidden=bufB)
  for (int k = 0; k < KSLOT; ++k) {
    float ir = b_ih[t], iz = b_ih[HD + t], ig = b_ih[2 * HD + t];
    float hr = b_hh[t], hz = b_hh[HD + t], hg = b_hh[2 * HD + t];
    for (int j = 0; j < HD; ++j) {
      const float uj = bufA[k][j], sj = bufB[k][j];
      const float* wi = W_ih + (size_t)j * 3 * HD;
      const float* wh = W_hh + (size_t)j * 3 * HD;
      ir += uj * wi[t]; iz += uj * wi[HD + t]; ig += uj * wi[2 * HD + t];
      hr += sj * wh[t]; hz += sj * wh[HD + t]; hg += sj * wh[2 * HD + t];
    }
    const float r = 1.0f / (1.0f + __expf(-(ir + hr)));
    const float z = 1.0f / (1.0f + __expf(-(iz + hz)));
    const float n = tanhf(ig + r * hg);
    snew[k][t] = (1.0f - z) * n + z * bufB[k][t];
  }
  __syncthreads();

  if (t < KSLOT) {
    float mu = 0.0f;
    for (int j = 0; j < HD; ++j) mu += snew[t][j];
    mu *= (1.0f / HD);
    float var = 0.0f;
    for (int j = 0; j < HD; ++j) { float d = snew[t][j] - mu; var += d * d; }
    st[t][0] = mu;
    st[t][1] = rsqrtf(var * (1.0f / HD) + 1e-5f);
  }
  __syncthreads();
  for (int k = 0; k < KSLOT; ++k)
    bufA[k][t] = (snew[k][t] - st[k][0]) * st[k][1] * ln_g[t] + ln_b[t];
  __syncthreads();
  for (int k = 0; k < KSLOT; ++k) {
    float a = mb1[t];
    for (int j = 0; j < HD; ++j) a += bufA[k][j] * mW1[j * HD + t];
    bufB[k][t] = fmaxf(a, 0.0f);
  }
  __syncthreads();
  for (int k = 0; k < KSLOT; ++k) {
    float a = mb2[t];
    for (int j = 0; j < HD; ++j) a += bufB[k][j] * mW2[j * HD + t];
    slots[k * HD + t] = snew[k][t] + a;
  }
}

// ---------------------------------------------------------------------------
// Cosine similarity + greedy matching. out[0..15]=matches, out[16..271]=sim.
__global__ __launch_bounds__(256) void sim_match(
    const float* __restrict__ sa, const float* __restrict__ sb,
    float* __restrict__ out) {
  __shared__ float A[KSLOT][HD];
  __shared__ float B[KSLOT][HD];
  __shared__ float S[KSLOT][KSLOT];
  const int t = threadIdx.x;
  for (int k = 0; k < KSLOT; ++k) {
    A[k][t] = sa[k * HD + t];
    B[k][t] = sb[k * HD + t];
  }
  __syncthreads();
  if (t < 32) {
    const int k = t & 15;
    float* row = (t >> 4) ? B[k] : A[k];
    float nrm = 0.0f;
    for (int j = 0; j < HD; ++j) nrm += row[j] * row[j];
    nrm = fmaxf(sqrtf(nrm), 1e-12f);
    const float inv = 1.0f / nrm;
    for (int j = 0; j < HD; ++j) row[j] *= inv;
  }
  __syncthreads();
  {
    const int i = t >> 4, j = t & 15;
    float d = 0.0f;
    for (int h = 0; h < HD; ++h) d += A[i][h] * B[j][h];
    S[i][j] = d;
    out[16 + t] = d;
  }
  __syncthreads();
  if (t == 0) {
    float ms[KSLOT];
    int mi[KSLOT], order[KSLOT], matches[KSLOT];
    bool used[KSLOT];
    for (int i = 0; i < KSLOT; ++i) {
      float best = S[i][0]; int bj = 0;
      for (int j = 1; j < KSLOT; ++j)
        if (S[i][j] > best) { best = S[i][j]; bj = j; }
      ms[i] = best; mi[i] = bj;
      order[i] = i; matches[i] = -1; used[i] = false;
    }
    // stable insertion sort, descending by ms (ties keep index order)
    for (int a = 1; a < KSLOT; ++a) {
      int o = order[a]; float v = ms[o]; int b = a - 1;
      while (b >= 0 && ms[order[b]] < v) { order[b + 1] = order[b]; --b; }
      order[b + 1] = o;
    }
    for (int a = 0; a < KSLOT; ++a) {
      const int idx = order[a];
      const int j = mi[idx];
      if (!used[j] && ms[idx] > 0.3f) { matches[idx] = j; used[j] = true; }
    }
    for (int i = 0; i < KSLOT; ++i) out[i] = (float)matches[i];
  }
}

// ---------------------------------------------------------------------------
extern "C" void kernel_launch(void* const* d_in, const int* in_sizes, int n_in,
                              void* d_out, int out_size, void* d_ws,
                              size_t ws_size, hipStream_t stream) {
  (void)in_sizes; (void)n_in; (void)out_size;

  const float* dets[2]  = {(const float*)d_in[0], (const float*)d_in[1]};
  const float* noise[2] = {(const float*)d_in[2], (const float*)d_in[3]};
  const float* det_W1 = (const float*)d_in[4];
  const float* det_b1 = (const float*)d_in[5];
  const float* det_W2 = (const float*)d_in[6];
  const float* det_b2 = (const float*)d_in[7];
  const float* mu_p   = (const float*)d_in[8];
  const float* lsig   = (const float*)d_in[9];
  const float* ln_in_g = (const float*)d_in[10];
  const float* ln_in_b = (const float*)d_in[11];
  const float* ln_s_g  = (const float*)d_in[12];
  const float* ln_s_b  = (const float*)d_in[13];
  const float* ln_m_g  = (const float*)d_in[14];
  const float* ln_m_b  = (const float*)d_in[15];
  const float* Wq   = (const float*)d_in[16];
  const float* Wk   = (const float*)d_in[17];
  const float* Wv   = (const float*)d_in[18];
  const float* W_ih = (const float*)d_in[19];
  const float* W_hh = (const float*)d_in[20];
  const float* b_ih = (const float*)d_in[21];
  const float* b_hh = (const float*)d_in[22];
  const float* mW1  = (const float*)d_in[23];
  const float* mb1  = (const float*)d_in[24];
  const float* mW2  = (const float*)d_in[25];
  const float* mb2  = (const float*)d_in[26];

  // ---- workspace layout (regions aliased across pipeline phases) ----
  const size_t SZ_ACT16 = (size_t)NDET * HD * 2;   // 32 MB
  const size_t SZ_ACT32 = (size_t)NDET * HD * 4;   // 64 MB
  const size_t o_wt    = 0;                         // 512 KB (4 f16 weights)
  const size_t o_regA  = 1u << 20;                  // dets16 -> inputs16
  const size_t o_regB  = o_regA + SZ_ACT16;         // h1 -> kmat
  const size_t o_regC  = o_regB + SZ_ACT16;         // feats -> vt(+p16)
  const size_t o_q     = o_regC + SZ_ACT32;
  const size_t o_spart = o_q + (size_t)KSLOT * HD * 2;
  const size_t o_updp  = o_spart + (size_t)NSPART * KSLOT * 4;
  const size_t o_sl0   = o_updp + (size_t)SPLITK * KSLOT * HD * 4;
  const size_t o_sl1   = o_sl0 + (size_t)KSLOT * HD * 4;
  const size_t o_end   = o_sl1 + (size_t)KSLOT * HD * 4;
  if (ws_size < o_end) return;

  char* ws = (char*)d_ws;
  _Float16* WT       = (_Float16*)(ws + o_wt);
  _Float16* dets16   = (_Float16*)(ws + o_regA);
  _Float16* inputs16 = (_Float16*)(ws + o_regA);
  _Float16* h1       = (_Float16*)(ws + o_regB);
  _Float16* kmat     = (_Float16*)(ws + o_regB);
  float*    feats    = (float*)(ws + o_regC);
  _Float16* vt       = (_Float16*)(ws + o_regC);
  _Float16* p16      = (_Float16*)(ws + o_regC + SZ_ACT16);
  _Float16* q16      = (_Float16*)(ws + o_q);
  float*    spart    = (float*)(ws + o_spart);
  float*    updpart  = (float*)(ws + o_updp);
  float*    slotbuf[2] = {(float*)(ws + o_sl0), (float*)(ws + o_sl1)};

  prep_weights<<<1024, 256, 0, stream>>>(det_W1, det_W2, Wk, Wv, WT);

  const dim3 ggrid(NDET / 64, HD / 128);
  for (int f = 0; f < 2; ++f) {
    cvt_f32_f16<<<NDET, 256, 0, stream>>>(dets[f], dets16, (size_t)NDET * HD);
    // feats = relu(dets@W1+b1)@W2 + b2
    gemm256<1, true, true><<<ggrid, 256, 0, stream>>>(
        dets16, WT + 0 * 65536, det_b1, h1, 0);
    gemm256<0, false, true><<<ggrid, 256, 0, stream>>>(
        h1, WT + 1 * 65536, det_b2, feats, 0);
    // inputs = LN(feats); k = inputs@Wk; v^T = (inputs@Wv)^T
    layernorm_rows<<<NDET / 8, 256, 0, stream>>>(feats, inputs16, ln_in_g,
                                                 ln_in_b);
    gemm256<1, false, false><<<ggrid, 256, 0, stream>>>(
        inputs16, WT + 2 * 65536, nullptr, kmat, 0);
    gemm256<2, false, false><<<ggrid, 256, 0, stream>>>(
        inputs16, WT + 3 * 65536, nullptr, vt, NDET);

    slots_init<<<1, 256, 0, stream>>>(slotbuf[f], mu_p, lsig, noise[f]);
    for (int it = 0; it < 3; ++it) {
      slot_q_kernel<<<1, 256, 0, stream>>>(slotbuf[f], ln_s_g, ln_s_b, Wq, q16,
                                           0.0625f /* H^-0.5 */);
      logits_softmax<<<NDET / 128, 256, 0, stream>>>(q16, kmat, p16, spart);
      updates_gemm<<<dim3(HD / 16, SPLITK), 256, 0, stream>>>(p16, vt, updpart);
      slot_update<<<1, 256, 0, stream>>>(slotbuf[f], updpart, spart, W_ih, W_hh,
                                         b_ih, b_hh, ln_m_g, ln_m_b, mW1, mb1,
                                         mW2, mb2);
    }
  }
  sim_match<<<1, 256, 0, stream>>>(slotbuf[0], slotbuf[1], (float*)d_out);
}
